// VectorQuantizer_24739011625475
// MI455X (gfx1250) — matrix operand order repaired and checked by reading, not compiled, not run on hardware
//
#include <hip/hip_runtime.h>

#define B_N 32768
#define D_N 256
#define K_N 4096
#define BETA 0.25f

typedef __attribute__((ext_vector_type(16))) __bf16   v16bf;
typedef __attribute__((ext_vector_type(8)))  float    v8f;
typedef __attribute__((ext_vector_type(4)))  unsigned v4u;   // asm-friendly 128b

// LDS tile rows padded to 272 bf16 (544B stride, 32B aligned, breaks the
// 16-way bank conflict a 512B stride would cause on ds_load_b128)
#define TROW 272
#define TILE_ROWS 32            // 2 N-tiles of 16 codebook rows per buffer

// workspace layout (bytes)
static const size_t WS_CB     = 0;                                   // K*D bf16
static const size_t WS_WNORM  = (size_t)K_N * D_N * 2;               // K f32
static const size_t WS_COUNTS = WS_WNORM + (size_t)K_N * 4;          // K u32
static const size_t WS_IDX    = WS_COUNTS + (size_t)K_N * 4;         // B i32
static const size_t WS_SSE    = WS_IDX + (size_t)B_N * 4;            // 1 f32

__device__ __forceinline__ float waveReduceSum(float v) {
#pragma unroll
    for (int off = 16; off > 0; off >>= 1)
        v += __shfl_xor(v, off, 32);
    return v;
}

// ---------------- prep: codebook f32 -> bf16, row norms, zero counts/sse ---
__global__ void vq_prep_kernel(const float* __restrict__ cb,
                               unsigned short* __restrict__ cb_bf,
                               float* __restrict__ wnorm,
                               unsigned int* __restrict__ counts,
                               float* __restrict__ sse) {
    const int row = blockIdx.x;
    const int t   = threadIdx.x;
    float w = cb[(size_t)row * D_N + t];
    __bf16 wb = (__bf16)w;
    cb_bf[(size_t)row * D_N + t] = __builtin_bit_cast(unsigned short, wb);

    float s = waveReduceSum(w * w);
    __shared__ float tmp[8];
    const int wv = t >> 5, lane = t & 31;
    if (lane == 0) tmp[wv] = s;
    __syncthreads();
    if (wv == 0) {
        float v = (lane < 8) ? tmp[lane] : 0.0f;
        v = waveReduceSum(v);
        if (lane == 0) {
            wnorm[row]  = v;
            counts[row] = 0u;
            if (row == 0) *sse = 0.0f;
        }
    }
}

// B fragment = 32B of LDS (16 bf16) = two ds_load_b128
union Frag {
    v4u   q[2];
    v16bf v;
};

// issue both halves of a fragment; program-order preserved (volatile asm)
#define LDS_LOAD_FRAG(F, ADDR)                                         \
    asm volatile("ds_load_b128 %0, %2\n\t"                             \
                 "ds_load_b128 %1, %2 offset:16"                       \
                 : "=v"((F).q[0]), "=v"((F).q[1]) : "v"(ADDR))

// partial dscnt wait. Reads the guarded fragments (wait cannot rise above
// their loads) and passes one scalar token of each accumulator through a
// tied 32-bit operand: the WMMAs consuming the accumulators therefore cannot
// be scheduled before this wait.
#define DS_WAIT_GUARD(F0, F1, ACC0, ACC1, IMM)                         \
    do {                                                               \
        float _t0 = (ACC0)[0];                                         \
        float _t1 = (ACC1)[0];                                         \
        asm volatile("s_wait_dscnt " IMM                               \
                     : "+v"(_t0), "+v"(_t1)                            \
                     : "v"((F0).q[0]), "v"((F0).q[1]),                 \
                       "v"((F1).q[0]), "v"((F1).q[1]));                \
        (ACC0)[0] = _t0;                                               \
        (ACC1)[0] = _t1;                                               \
    } while (0)

// ---------------- search: async-LDS staged codebook + WMMA + argmin --------
__global__ void __launch_bounds__(256)
vq_search_kernel(const float* __restrict__ z_e,
                 const unsigned short* __restrict__ cb_bf,
                 const float* __restrict__ wnorm,
                 unsigned int* __restrict__ counts,
                 int* __restrict__ idx_buf,
                 float* __restrict__ out_idx) {
    __shared__ __align__(32) unsigned short lds_tile[2][TILE_ROWS * TROW];

    const int t    = threadIdx.x;
    const int lane = t & 31;
    const int wv   = t >> 5;
    const int half = lane >> 4;       // which 16-lane half
    const int mloc = lane & 15;
    const int m0   = blockIdx.x * 128 + wv * 16;   // 8 waves * 16 rows

    // ---- load A fragments (16 x 256 z tile, bf16, contiguous k order, same as B) ----
    // lane L (half h) element i: K = 32*c + 16h + i
    const float* zr = z_e + (size_t)(m0 + mloc) * D_N;
    v16bf a[8];
#pragma unroll
    for (int c = 0; c < 8; ++c) {
        const float* p0 = zr + 32 * c + 16 * half;
        const float* p1 = zr + 32 * c + 16 * half + 8;
#pragma unroll
        for (int j = 0; j < 8; ++j) {
            a[c][j]     = (__bf16)p0[j];
            a[c][8 + j] = (__bf16)p1[j];
        }
    }

    // cooperative tile copy: 16KB buffer = 1024 x 16B chunks; thread t owns
    // chunks t+q*256 (4 async b128 instructions per wave per buffer)
    auto issue_copy = [&](int kt2, int buf) {
#pragma unroll
        for (int q = 0; q < 4; ++q) {
            const int c   = t + q * 256;
            const int row = c >> 5;            // 32 chunks per 256-elem row
            const int col = (c & 31) * 8;
            const unsigned short* g = cb_bf + (size_t)kt2 * (TILE_ROWS * D_N)
                                            + row * D_N + col;
            unsigned short* l = &lds_tile[buf][row * TROW + col];
            unsigned lds_off = (unsigned)(unsigned long long)l;
            unsigned long long ga = (unsigned long long)g;
            asm volatile("global_load_async_to_lds_b128 %0, %1, off"
                         :: "v"(lds_off), "v"(ga) : "memory");
        }
    };

    float best[16];
    int   bidx[16];
#pragma unroll
    for (int r = 0; r < 16; ++r) { best[r] = 3.4e38f; bidx[r] = 0; }

    constexpr int NT2 = K_N / TILE_ROWS;       // 128 iterations of 32 rows
    issue_copy(0, 0);                          // prologue: buffer 0 in flight

#pragma unroll 1
    for (int kt2 = 0; kt2 < NT2; ++kt2) {
        const int cur = kt2 & 1;
        if (kt2 + 1 < NT2) {
            issue_copy(kt2 + 1, cur ^ 1);      // keep next buffer in flight
            asm volatile("s_wait_asynccnt 0x4" ::: "memory");  // cur done
        } else {
            asm volatile("s_wait_asynccnt 0x0" ::: "memory");
        }
        __syncthreads();                       // cur buffer visible to all

        // Two N-tiles share the A fragments. B fragments flow through an
        // asm-pinned 2-chunk lookahead window: chunk c+2 loads are issued
        // before chunk c's WMMAs; partial s_wait_dscnt keeps 8 ds_loads in
        // flight while WMMAs execute.
        const unsigned short* wr0 = &lds_tile[cur][mloc * TROW + 16 * half];
        const unsigned base0 = (unsigned)(unsigned long long)wr0;
        const unsigned tile1 = 16 * TROW * 2;  // byte offset to 2nd N-tile

        Frag f0[8], f1[8];
#define LOADPAIR(C)                                                    \
        do { LDS_LOAD_FRAG(f0[(C)], base0 + 64 * (C));                 \
             LDS_LOAD_FRAG(f1[(C)], base0 + tile1 + 64 * (C)); } while (0)

        LOADPAIR(0);
        LOADPAIR(1);

        v8f acc0 = {0.f, 0.f, 0.f, 0.f, 0.f, 0.f, 0.f, 0.f};
        v8f acc1 = {0.f, 0.f, 0.f, 0.f, 0.f, 0.f, 0.f, 0.f};

#define STEP(C, WIMM)                                                  \
        do {                                                           \
            if ((C) + 2 <= 7) LOADPAIR((C) + 2);                       \
            DS_WAIT_GUARD(f0[(C)], f1[(C)], acc0, acc1, WIMM);         \
            acc0 = __builtin_amdgcn_wmma_f32_16x16x32_bf16(            \
                false, a[(C)], false, f0[(C)].v, (short)0, acc0,       \
                false, false);                                         \
            acc1 = __builtin_amdgcn_wmma_f32_16x16x32_bf16(            \
                false, a[(C)], false, f1[(C)].v, (short)0, acc1,       \
                false, false);                                         \
        } while (0)

        STEP(0, "0x8"); STEP(1, "0x8"); STEP(2, "0x8"); STEP(3, "0x8");
        STEP(4, "0x8"); STEP(5, "0x8"); STEP(6, "0x4"); STEP(7, "0x0");
#undef STEP
#undef LOADPAIR

        const int   n0  = kt2 * TILE_ROWS + mloc;
        const int   n1  = n0 + 16;
        const float wn0 = wnorm[n0];
        const float wn1 = wnorm[n1];
#pragma unroll
        for (int r = 0; r < 8; ++r) {
            float dv0 = wn0 - 2.0f * acc0[r];  // ||w||^2 - 2 z.w (argmin-equiv)
            float dv1 = wn1 - 2.0f * acc1[r];
            if (dv0 < best[r])     { best[r]     = dv0; bidx[r]     = n0; }
            if (dv1 < best[8 + r]) { best[8 + r] = dv1; bidx[8 + r] = n1; }
        }

        __syncthreads();                 // all reads of buf[cur] done before
    }                                    // it is overwritten at kt2+1's issue

    // merge the two N-tile argmin streams (n0 < n1 everywhere: keep ties low)
#pragma unroll
    for (int r = 0; r < 8; ++r) {
        if (best[8 + r] < best[r] ||
            (best[8 + r] == best[r] && bidx[8 + r] < bidx[r])) {
            best[r] = best[8 + r];
            bidx[r] = bidx[8 + r];
        }
    }

    // ---- reduce argmin across the 16 lanes of each half ----
#pragma unroll
    for (int r = 0; r < 8; ++r) {
        float bv = best[r];
        int   bi = bidx[r];
#pragma unroll
        for (int off = 8; off > 0; off >>= 1) {
            float ov = __shfl_xor(bv, off, 32);
            int   oi = __shfl_xor(bi, off, 32);
            if (ov < bv || (ov == bv && oi < bi)) { bv = ov; bi = oi; }
        }
        if (mloc == 0) {
            const int row = m0 + 8 * half + r;   // C layout: M = r + 8*half
            idx_buf[row] = bi;
            out_idx[row] = (float)bi;
            atomicAdd(&counts[bi], 1u);
        }
    }
}

// ---------------- gather z_q + accumulate squared error --------------------
__global__ void vq_gather_kernel(const float* __restrict__ z_e,
                                 const float* __restrict__ cb,
                                 const int* __restrict__ idx_buf,
                                 float* __restrict__ zq_out,
                                 float* __restrict__ sse) {
    const int row = blockIdx.x;
    const int t   = threadIdx.x;
    const int idx = idx_buf[row];
    float w = cb[(size_t)idx * D_N + t];
    float z = z_e[(size_t)row * D_N + t];
    zq_out[(size_t)row * D_N + t] = w;   // z_q_st == z_q in forward value
    float d = w - z;
    float s = waveReduceSum(d * d);
    __shared__ float tmp[8];
    const int wv = t >> 5, lane = t & 31;
    if (lane == 0) tmp[wv] = s;
    __syncthreads();
    if (wv == 0) {
        float v = (lane < 8) ? tmp[lane] : 0.0f;
        v = waveReduceSum(v);
        if (lane == 0) atomicAdd(sse, v);
    }
}

// ---------------- finalize: perplexity + quant loss ------------------------
__global__ void vq_finalize_kernel(const unsigned int* __restrict__ counts,
                                   const float* __restrict__ sse,
                                   float* __restrict__ out_scalars) {
    const int t = threadIdx.x;
    float local = 0.0f;
    for (int k = t; k < K_N; k += 256) {
        float p = (float)counts[k] * (1.0f / (float)B_N);
        local += p * __logf(p + 1e-10f);
    }
    float s = waveReduceSum(local);
    __shared__ float tmp[8];
    const int wv = t >> 5, lane = t & 31;
    if (lane == 0) tmp[wv] = s;
    __syncthreads();
    if (t == 0) {
        float h = 0.0f;
        for (int i = 0; i < 8; ++i) h += tmp[i];
        // quant_loss = (1 + beta) * mean((z_q - z_e)^2)
        out_scalars[0] = (*sse) * ((1.0f + BETA) / ((float)B_N * (float)D_N));
        out_scalars[1] = __expf(-h);
    }
}

extern "C" void kernel_launch(void* const* d_in, const int* in_sizes, int n_in,
                              void* d_out, int out_size, void* d_ws, size_t ws_size,
                              hipStream_t stream) {
    (void)in_sizes; (void)n_in; (void)out_size; (void)ws_size;
    const float* z_e = (const float*)d_in[0];
    const float* cbw = (const float*)d_in[1];
    float* out = (float*)d_out;
    char*  ws  = (char*)d_ws;

    unsigned short* cb_bf  = (unsigned short*)(ws + WS_CB);
    float*          wnorm  = (float*)(ws + WS_WNORM);
    unsigned int*   counts = (unsigned int*)(ws + WS_COUNTS);
    int*            idxb   = (int*)(ws + WS_IDX);
    float*          sse    = (float*)(ws + WS_SSE);

    float* out_zq   = out;                                  // [B, D]
    float* out_idx  = out + (size_t)B_N * D_N;              // [B] (as float)
    float* out_scal = out + (size_t)B_N * D_N + B_N;        // loss, perplexity

    vq_prep_kernel<<<K_N, 256, 0, stream>>>(cbw, cb_bf, wnorm, counts, sse);
    vq_search_kernel<<<B_N / 128, 256, 0, stream>>>(z_e, cb_bf, wnorm, counts,
                                                    idxb, out_idx);
    vq_gather_kernel<<<B_N, 256, 0, stream>>>(z_e, cbw, idxb, out_zq, sse);
    vq_finalize_kernel<<<1, 256, 0, stream>>>(counts, sse, out_scal);
}
